// Convolution_51402168599159
// MI455X (gfx1250) — matrix operand order repaired and checked by reading, not compiled, not run on hardware
//
#include <hip/hip_runtime.h>
#include <hip/hip_bf16.h>

// TFN convolution, fully fused, f16 WMMA radial networks on gfx1250 (wave32),
// with the image stream fed by the Tensor Data Mover (async tensor -> LDS,
// double buffered, TENSORcnt-synchronized).
//
// Shapes: B=4, N=512, RBF=32, F=16.
// One wave per (m, a) output row. Loop over 32 b-tiles of 16 pairs each.
//   TDM: image[m, a, b0:b0+16, 0:32] (16x32 f32 tile of a contiguous 512x32
//        slice) -> LDS, padded 16B per 128B row (bank-conflict-free reads).
//   A  = f16(tile)  (16x32, M=pair, K=rbf)   from LDS ds_load_b128
//   Bw = f16(W)     (32x16, K=rbf, N=f)      preloaded per wave
//   D  = v_wmma_f32_16x16x32_f16(A, Bw, C=bias-splat), one per W00/W01/W10/W11
// Epilogue consumes D in-register (lane = f, vgpr = pair row), applies the
// self-pair mask, accumulates out1..out5 (Levi-Civita term = cross(v, k)).
// r00/r01/r10/r11 (67 MB each) are never materialized: HBM traffic ~150 MB.

typedef __attribute__((ext_vector_type(16))) _Float16     v16h;
typedef __attribute__((ext_vector_type(8)))  float        v8f;
typedef __attribute__((ext_vector_type(4)))  unsigned int v4u;
typedef __attribute__((ext_vector_type(8)))  int          v8i;
typedef __attribute__((ext_vector_type(4)))  int          v4i;

#define BB   4
#define NN   512
#define RBFD 32
#define FF   16
#define WAVES_PER_BLOCK 8

#define TILE_ROWS   16
#define ROW_BYTES   128                       // 32 f32
#define PAD_BYTES   16                        // TDM pad: 4 dwords per 32 dwords
#define TILE_STRIDE (ROW_BYTES + PAD_BYTES)   // 144 B LDS row stride
#define TILE_LDS    (TILE_ROWS * TILE_STRIDE) // 2304 B per tile buffer

// output tuple offsets (flat f32 elements)
#define O1 0                      // (B,N,F)    32768
#define O2 32768                  // (B,N,F,3)  98304
#define O3 131072                 // (B,N,F,3)  98304
#define O4 229376                 // (B,N,F)    32768
#define O5 262144                 // (B,N,F,3)  98304

// Issue one TDM descriptor: 16x32-f32 tile starting at `src` (rows contiguous,
// stride 32 f32) into LDS byte offset `lds_off`, with 4-dword padding after
// every 32 stored dwords (-> 144B LDS row stride).
__device__ __forceinline__ void tdm_load_tile(unsigned lds_off, const float* src) {
  unsigned long long ga = (unsigned long long)(const void*)src;
  v4u g0;
  g0[0] = 1u;                                            // count=1, user desc
  g0[1] = lds_off;                                       // lds_addr
  g0[2] = (unsigned)ga;                                  // global_addr[31:0]
  g0[3] = (unsigned)((ga >> 32) & 0x01FFFFFFu)           // global_addr[56:32]
        | (2u << 30);                                    // type = 2 (image)
  v8i g1;
  g1[0] = (int)((2u << 16)        // data_size = 4B
              | (1u << 20)        // pad_enable
              | (4u << 22)        // pad_interval: 2^(4+1)=32 dwords
              | (3u << 25));      // pad_amount: 3+1 = 4 dwords
  g1[1] = (int)(32u << 16);       // tensor_dim0 = 32   (row length)
  g1[2] = (int)(512u << 16);      // tensor_dim1 = 512  (rows)
  g1[3] = (int)(32u << 16);       // tile_dim0 = 32
  g1[4] = 16;                     // tile_dim1 = 16
  g1[5] = 32;                     // tensor_dim0_stride = 32
  g1[6] = 0;
  g1[7] = 0;
  v4i gz = {0, 0, 0, 0};
#if defined(__clang_major__) && (__clang_major__ >= 23)
  v8i gz8 = {0, 0, 0, 0, 0, 0, 0, 0};
  __builtin_amdgcn_tensor_load_to_lds(g0, g1, gz, gz, gz8, 0);
#else
  __builtin_amdgcn_tensor_load_to_lds(g0, g1, gz, gz, 0);
#endif
}

__global__ __launch_bounds__(256) void tfn_conv_kernel(
    const float* __restrict__ image,
    const float* __restrict__ vecs,
    const float* __restrict__ feat0,
    const float* __restrict__ feat1,
    const float* __restrict__ W00, const float* __restrict__ b00,
    const float* __restrict__ W01, const float* __restrict__ b01,
    const float* __restrict__ W10, const float* __restrict__ b10,
    const float* __restrict__ W11, const float* __restrict__ b11,
    float* __restrict__ out)
{
  __shared__ __align__(16) unsigned char smem[WAVES_PER_BLOCK * 2 * TILE_LDS];

  const int lane = threadIdx.x & 31;
  const int wave = threadIdx.x >> 5;
  const int ag   = blockIdx.x * WAVES_PER_BLOCK + wave;  // 0 .. B*N-1
  const int m    = ag >> 9;           // batch
  const int a    = ag & (NN - 1);     // output row
  const int f    = lane & 15;         // feature owned by this lane (D-layout N)
  const int hi   = lane >> 4;         // wave half

  // ---- radial weights -> WMMA B-operand layout (v16h per lane) ----
  // 16-bit B 32x16: vgpr v holds K = {2v, 2v+1} (+8 for v >= 4) for lanes 0-15
  // (K += 8 for lanes 16-31), N = lane. So half h of this lane =
  // W[h < 8 ? 8*hi + h : 16 + 8*hi + (h - 8)][f].
  v16h wb00, wb01, wb10, wb11;
#pragma unroll
  for (int h = 0; h < 16; ++h) {
    const int k = (h < 8) ? (8 * hi + h) : (16 + 8 * hi + (h - 8));
    wb00[h] = (_Float16)W00[k * FF + f];
    wb01[h] = (_Float16)W01[k * FF + f];
    wb10[h] = (_Float16)W10[k * FF + f];
    wb11[h] = (_Float16)W11[k * FF + f];
  }
  // Biases are constant per lane (lane = f in the C/D layout): fold them into
  // the WMMA C operand instead of adding per element in the epilogue.
  const float bias00 = b00[f], bias01 = b01[f];
  const float bias10 = b10[f], bias11 = b11[f];
  const v8f c00 = {bias00, bias00, bias00, bias00, bias00, bias00, bias00, bias00};
  const v8f c01 = {bias01, bias01, bias01, bias01, bias01, bias01, bias01, bias01};
  const v8f c10 = {bias10, bias10, bias10, bias10, bias10, bias10, bias10, bias10};
  const v8f c11 = {bias11, bias11, bias11, bias11, bias11, bias11, bias11, bias11};

  // ---- per-(a,f) accumulators over all pairs b ----
  float acc1 = 0.f, acc4 = 0.f;
  float a2x = 0.f, a2y = 0.f, a2z = 0.f;
  float a3x = 0.f, a3y = 0.f, a3z = 0.f;
  float a5x = 0.f, a5y = 0.f, a5z = 0.f;

  const float* imbase = image + (size_t)(m * NN + a) * NN * RBFD;
  const float* vbase  = vecs  + (size_t)(m * NN + a) * NN * 3;

  const unsigned lds_base   = (unsigned)(size_t)(const void*)smem
                            + (unsigned)(wave * 2 * TILE_LDS);
  const unsigned char* tbase = smem + wave * 2 * TILE_LDS;

  // ---- TDM prologue: fetch tile 0 into buffer 0 ----
  tdm_load_tile(lds_base, imbase);

  for (int bt = 0; bt < NN / 16; ++bt) {
    const int cur = bt & 1;
    if (bt + 1 < NN / 16) {
      // Buffer (bt+1)&1 was last read in the previous iteration; make sure our own
      // LDS reads have drained before the TDM overwrites it, then prefetch.
      asm volatile("s_wait_dscnt 0" ::: "memory");
      tdm_load_tile(lds_base + (unsigned)(((bt + 1) & 1) * TILE_LDS),
                    imbase + (size_t)(bt + 1) * 16 * RBFD);
      __builtin_amdgcn_s_wait_tensorcnt(1);   // tile bt done (in-order TDM)
    } else {
      __builtin_amdgcn_s_wait_tensorcnt(0);   // last tile done
    }

    // ---- A operand from LDS: 16-bit A layout ----
    // lane: row = lane&15, K chunks [8*hi, 8*hi+8) and [16+8*hi, +8).
    // LDS row stride 144B (TDM pad) => conflict-free ds_load_b128.
    const unsigned char* tile = tbase + cur * TILE_LDS;
    const unsigned char* rowp = tile + (size_t)f * TILE_STRIDE + 32 * hi;
    const float4 q0 = *reinterpret_cast<const float4*>(rowp);
    const float4 q1 = *reinterpret_cast<const float4*>(rowp + 16);
    const float4 q2 = *reinterpret_cast<const float4*>(rowp + 64);
    const float4 q3 = *reinterpret_cast<const float4*>(rowp + 80);
    v16h A;
    A[0]  = (_Float16)q0.x; A[1]  = (_Float16)q0.y;
    A[2]  = (_Float16)q0.z; A[3]  = (_Float16)q0.w;
    A[4]  = (_Float16)q1.x; A[5]  = (_Float16)q1.y;
    A[6]  = (_Float16)q1.z; A[7]  = (_Float16)q1.w;
    A[8]  = (_Float16)q2.x; A[9]  = (_Float16)q2.y;
    A[10] = (_Float16)q2.z; A[11] = (_Float16)q2.w;
    A[12] = (_Float16)q3.x; A[13] = (_Float16)q3.y;
    A[14] = (_Float16)q3.z; A[15] = (_Float16)q3.w;

    // ---- four radial projections for this 16-pair tile (bias in C) ----
    v8f d00 = __builtin_amdgcn_wmma_f32_16x16x32_f16(false, A, false, wb00,
                                                     (short)0, c00, false, false);
    v8f d01 = __builtin_amdgcn_wmma_f32_16x16x32_f16(false, A, false, wb01,
                                                     (short)0, c01, false, false);
    v8f d10 = __builtin_amdgcn_wmma_f32_16x16x32_f16(false, A, false, wb10,
                                                     (short)0, c10, false, false);
    v8f d11 = __builtin_amdgcn_wmma_f32_16x16x32_f16(false, A, false, wb11,
                                                     (short)0, c11, false, false);

    // ---- epilogue: D layout row v  <->  pair b = bt*16 + v + 8*hi ----
#pragma unroll
    for (int v = 0; v < 8; ++v) {
      const int b = bt * 16 + v + 8 * hi;
      const float vx = vbase[(size_t)b * 3 + 0];   // uniform per half-wave
      const float vy = vbase[(size_t)b * 3 + 1];
      const float vz = vbase[(size_t)b * 3 + 2];
      const float n2 = vx * vx + vy * vy + vz * vz;
      const bool keep = n2 >= 1e-14f;              // (EPS=1e-7)^2

      const size_t fb = (size_t)(m * NN + b) * FF + f;
      const float f0v = feat0[fb];                 // coalesced across lanes
      const float k0 = feat1[fb * 3 + 0];
      const float k1 = feat1[fb * 3 + 1];
      const float k2 = feat1[fb * 3 + 2];

      const float r00v = d00[v];                   // bias already in C
      const float r01v = keep ? d01[v] : 0.f;
      const float r10v = d10[v];
      const float r11v = keep ? d11[v] : 0.f;

      acc1 += r00v * f0v;                          // 0x0 -> 0
      const float p = r01v * f0v;                  // 1x0 -> 1
      a2x += p * vx; a2y += p * vy; a2z += p * vz;
      a3x += r10v * k0; a3y += r10v * k1; a3z += r10v * k2;   // 0x1 -> 1
      acc4 += r11v * (vx * k0 + vy * k1 + vz * k2);           // 1x1 -> 0
      const float cx = vy * k2 - vz * k1;          // levi-civita: cross(v, k)
      const float cy = vz * k0 - vx * k2;
      const float cz = vx * k1 - vy * k0;
      a5x += r11v * cx; a5y += r11v * cy; a5z += r11v * cz;   // 1x1 -> 1
    }
  }

  // ---- cross-half reduction (rows 0-7 live in lanes 0-15, rows 8-15 in
  //      lanes 16-31 for the same f): xor-shuffle by 16 on wave32 ----
  acc1 += __shfl_xor(acc1, 16, 32);
  acc4 += __shfl_xor(acc4, 16, 32);
  a2x += __shfl_xor(a2x, 16, 32); a2y += __shfl_xor(a2y, 16, 32);
  a2z += __shfl_xor(a2z, 16, 32);
  a3x += __shfl_xor(a3x, 16, 32); a3y += __shfl_xor(a3y, 16, 32);
  a3z += __shfl_xor(a3z, 16, 32);
  a5x += __shfl_xor(a5x, 16, 32); a5y += __shfl_xor(a5y, 16, 32);
  a5z += __shfl_xor(a5z, 16, 32);

  if (hi == 0) {
    const size_t o = (size_t)(m * NN + a) * FF + f;   // (B,N,F) flat
    out[O1 + o] = acc1;
    out[O4 + o] = acc4;
    float* o2 = out + O2 + o * 3;
    o2[0] = a2x; o2[1] = a2y; o2[2] = a2z;
    float* o3 = out + O3 + o * 3;
    o3[0] = a3x; o3[1] = a3y; o3[2] = a3z;
    float* o5 = out + O5 + o * 3;
    o5[0] = a5x; o5[1] = a5y; o5[2] = a5z;
  }
}

extern "C" void kernel_launch(void* const* d_in, const int* in_sizes, int n_in,
                              void* d_out, int out_size, void* d_ws, size_t ws_size,
                              hipStream_t stream) {
  const float* image = (const float*)d_in[0];
  const float* vecs  = (const float*)d_in[1];
  const float* feat0 = (const float*)d_in[2];
  const float* feat1 = (const float*)d_in[3];
  const float* W00 = (const float*)d_in[4];
  const float* b00 = (const float*)d_in[5];
  const float* W01 = (const float*)d_in[6];
  const float* b01 = (const float*)d_in[7];
  const float* W10 = (const float*)d_in[8];
  const float* b10 = (const float*)d_in[9];
  const float* W11 = (const float*)d_in[10];
  const float* b11 = (const float*)d_in[11];

  const int total_waves = BB * NN;                       // 2048 a-rows
  dim3 grid(total_waves / WAVES_PER_BLOCK);              // 256 blocks
  dim3 block(WAVES_PER_BLOCK * 32);                      // 8 waves (wave32)
  tfn_conv_kernel<<<grid, block, 0, stream>>>(
      image, vecs, feat0, feat1,
      W00, b00, W01, b01, W10, b10, W11, b11,
      (float*)d_out);
}
